// CellWiseLSTM_41927470743580
// MI455X (gfx1250) — matrix operand order repaired and checked by reading, not compiled, not run on hardware
//
#include <hip/hip_runtime.h>
#include <hip/hip_bf16.h>
#include <string.h>

// Problem constants (match reference)
constexpr int B = 64, S = 512, E = 1024, H = 1024, L = 2;
constexpr int KDIM = 1024;            // E == H == 1024
constexpr int KT = KDIM / 32;         // 32 k-tiles of 32
constexpr int NTG_TOT = 4 * H / 16;   // 256 column tiles across all 4 gates
// Packed weight per layer (shorts): 256 ntg * 32 kt * 32 lanes * 16 halfs
constexpr size_t WPACK_LAYER = (size_t)NTG_TOT * KT * 32 * 16;   // = 4*H*KDIM
// Packed activation matrix [64 x 1024] in A-fragment order (shorts):
//   Apack[rt=4][kt=32][lane=32][16 halfs]
constexpr size_t APACK_MAT = (size_t)4 * KT * 32 * 16;           // 65536 shorts = 128KB

typedef __attribute__((ext_vector_type(16))) __bf16 v16bf;
typedef __attribute__((ext_vector_type(8)))  float  v8f;

__device__ __forceinline__ unsigned short f2bf(float f) {
  unsigned int u = __float_as_uint(f);
  unsigned int r = u + 0x7FFFu + ((u >> 16) & 1u); // round-to-nearest-even
  return (unsigned short)(r >> 16);
}

__device__ __forceinline__ float sigm(float x) {
  return 1.0f / (1.0f + __expf(-x));
}

// ---------------------------------------------------------------------------
// Pack W [4H x K] f32 (row-major over k) into WMMA B-fragment order, bf16:
//   P[ntg][kt][lane][16]  where lane's N = ntg*16 + (lane&15),
//   K base = kt*32 + (lane>=16 ? 8 : 0); halfs 0..7 -> K base+0..7, halfs 8..15 -> K base+16..23.
// ---------------------------------------------------------------------------
__global__ void lstm_pack_w(const float* __restrict__ W, unsigned short* __restrict__ P) {
  int idx = blockIdx.x * blockDim.x + threadIdx.x;      // one thread per lane-fragment
  int total = NTG_TOT * KT * 32;
  if (idx >= total) return;
  int lane = idx & 31;
  int kt   = (idx >> 5) % KT;
  int ntg  = (idx >> 5) / KT;
  int n  = ntg * 16 + (lane & 15);
  int kb = kt * 32 + ((lane >> 4) ? 8 : 0);
  const float* src = W + (size_t)n * KDIM + kb;
  unsigned short* dst = P + (size_t)idx * 16;
  #pragma unroll
  for (int j = 0; j < 16; ++j) dst[j] = f2bf(src[j < 8 ? j : j + 8]);
}

// ---------------------------------------------------------------------------
// Pack activation matrices [64 x 1024] f32 into A-fragment bf16 order.
// Element (m, k): rt=m/16, ml=m%16, kt=k/32, ko=k%32,
//   lane = ml + 16*((ko>>3)&1), half = (ko&7) + 8*(ko>=16)
// Each thread handles one k-octet (8 consecutive k): maps to 8 contiguous halfs.
// src element (mat, m, k) at src[mat*mat_stride + m*row_stride + k].
// ---------------------------------------------------------------------------
__global__ void lstm_pack_act(const float* __restrict__ src, unsigned short* __restrict__ dst,
                              long mat_stride, long row_stride, int nmat) {
  int idx = blockIdx.x * blockDim.x + threadIdx.x;
  int octs = KDIM / 8;                                  // 128 octets per row
  int total = nmat * B * octs;
  if (idx >= total) return;
  int oct = idx % octs;
  int m   = (idx / octs) % B;
  int mat = idx / (octs * B);
  int k   = oct * 8;
  int kt  = k >> 5;
  int sel = (k >> 3) & 3;                               // which octet within the k-tile
  int lane = (m & 15) + ((sel & 1) << 4);
  int hb   = (sel >> 1) << 3;                           // half base: 0 or 8
  const float* s = src + (size_t)mat * mat_stride + (size_t)m * row_stride + k;
  unsigned short* d = dst + (size_t)mat * APACK_MAT +
                      ((((size_t)(m >> 4) * KT + kt) * 32 + lane) << 4) + hb;
  #pragma unroll
  for (int j = 0; j < 8; ++j) d[j] = f2bf(s[j]);
}

__global__ void lstm_prep_bias(const float* __restrict__ bih, const float* __restrict__ bhh,
                               float* __restrict__ bias) {
  int i = blockIdx.x * blockDim.x + threadIdx.x;
  if (i < L * 4 * H) bias[i] = bih[i] + bhh[i];
}

__global__ void lstm_copy_c(const float* __restrict__ cx, float* __restrict__ cbuf) {
  int i = blockIdx.x * blockDim.x + threadIdx.x;
  if (i < L * B * H) cbuf[i] = cx[i];
}

// ---------------------------------------------------------------------------
// Fused LSTM step for one layer, one time step.
//   gates[64 x 4H] = x * Wih^T + h_prev * Whh^T + bias; then cell update.
// x and h_prev arrive PRE-PACKED in A-fragment bf16 order -> hot loop is
// pure vmem + wmma. Epilogue writes c (f32, in place), latest h (f32),
// h packed in A-fragment bf16 (for next step / next layer), optional output.
// Block = 128 threads (4 waves). Wave w owns batch rows 16w..16w+15.
// Block nt owns hidden columns nt*16..+15 for all four gates.
// ---------------------------------------------------------------------------
__global__ void __launch_bounds__(128)
lstm_step(const unsigned short* __restrict__ xpack,     // [4][KT][32][16] this step's x
          const unsigned short* __restrict__ hpack,     // [4][KT][32][16] h_prev this layer
          const unsigned short* __restrict__ wih,       // packed, this layer
          const unsigned short* __restrict__ whh,       // packed, this layer
          const float* __restrict__ bias,               // [4H] this layer
          float* __restrict__ cst,                      // [B,H] this layer (in place)
          float* __restrict__ hfin,                     // [B,H] latest h, f32
          unsigned short* __restrict__ hpack_out,       // packed h for next consumer
          float* __restrict__ outp, int out_row_stride) // d_out + t*H or null
{
  const int lane = threadIdx.x & 31;
  const int wv   = threadIdx.x >> 5;          // row tile 0..3
  const int nt   = blockIdx.x;                // column tile 0..63 (within H)
  const int nl   = lane & 15;                 // local column in C layout

  // Accumulators initialized with fused bias (bias per column == nl).
  v8f acc[4];
  #pragma unroll
  for (int g = 0; g < 4; ++g) {
    float bv = bias[g * H + nt * 16 + nl];
    #pragma unroll
    for (int r = 0; r < 8; ++r) acc[g][r] = bv;
  }

  // Per-kt stride in shorts: 32 lanes * 16 halfs.
  const int FRAG = 32 * 16;
  const unsigned short* xa = xpack + (((size_t)wv * KT) * 32 + lane) * 16;
  const unsigned short* ha = hpack + (((size_t)wv * KT) * 32 + lane) * 16;
  const unsigned short* wb[4];
  #pragma unroll
  for (int g = 0; g < 4; ++g)
    wb[g] = wih + ((((size_t)(g * 64 + nt)) * KT) * 32 + lane) * 16;

  // ---- x * Wih^T ----
  #pragma unroll 2
  for (int kt = 0; kt < KT; ++kt) {
    v16bf a = *reinterpret_cast<const v16bf*>(xa + (size_t)kt * FRAG);
    #pragma unroll
    for (int g = 0; g < 4; ++g) {
      const unsigned short* fp = wb[g] + (size_t)kt * FRAG;
      if (kt + 1 < KT) __builtin_prefetch(fp + FRAG, 0, 3);
      const v16bf b = *reinterpret_cast<const v16bf*>(fp);
      acc[g] = __builtin_amdgcn_wmma_f32_16x16x32_bf16(
          false, a, false, b, (short)0, acc[g], false, false);
    }
  }

  // ---- h_prev * Whh^T ----
  #pragma unroll
  for (int g = 0; g < 4; ++g)
    wb[g] = whh + ((((size_t)(g * 64 + nt)) * KT) * 32 + lane) * 16;
  #pragma unroll 2
  for (int kt = 0; kt < KT; ++kt) {
    v16bf a = *reinterpret_cast<const v16bf*>(ha + (size_t)kt * FRAG);
    #pragma unroll
    for (int g = 0; g < 4; ++g) {
      const unsigned short* fp = wb[g] + (size_t)kt * FRAG;
      if (kt + 1 < KT) __builtin_prefetch(fp + FRAG, 0, 3);
      const v16bf b = *reinterpret_cast<const v16bf*>(fp);
      acc[g] = __builtin_amdgcn_wmma_f32_16x16x32_bf16(
          false, a, false, b, (short)0, acc[g], false, false);
    }
  }

  // ---- cell update (C layout: VGPR r -> M = r + 8*(lane>=16), N = lane&15) ----
  const int ncol  = nt * 16 + nl;
  const int rbase = wv * 16 + ((lane >> 4) << 3);
  // Packed-h coordinates for column ncol (constant per thread):
  const int pkt   = ncol >> 5;
  const int pko   = ncol & 31;
  const int plane0 = ((pko >> 3) & 1) << 4;             // +16 if octet 1 or 3
  const int phalf  = (pko & 7) + ((pko >> 4) << 3);     // 0..15
  #pragma unroll
  for (int r = 0; r < 8; ++r) {
    int b = rbase + r;
    float gi = acc[0][r], gf = acc[1][r], gg = acc[2][r], go = acc[3][r];
    size_t off = (size_t)b * H + ncol;
    float cp = cst[off];
    float cn = sigm(gf) * cp + sigm(gi) * tanhf(gg);
    float hn = sigm(go) * tanhf(cn);
    cst[off]  = cn;
    hfin[off] = hn;
    // packed h: rt = b/16 == wv, ml = b%16
    int ml = b & 15;
    hpack_out[((((size_t)wv * KT + pkt) * 32 + (ml + plane0)) << 4) + phalf] = f2bf(hn);
    if (outp) outp[(size_t)b * out_row_stride + ncol] = hn;
  }
}

// ---------------------------------------------------------------------------
extern "C" void kernel_launch(void* const* d_in, const int* in_sizes, int n_in,
                              void* d_out, int out_size, void* d_ws, size_t ws_size,
                              hipStream_t stream) {
  const float* input_ = (const float*)d_in[0];  // [B,S,E]
  const float* hx     = (const float*)d_in[1];  // [L,B,H]
  const float* cx     = (const float*)d_in[2];  // [L,B,H]
  const float* W_ih   = (const float*)d_in[3];  // [L,4H,E]
  const float* W_hh   = (const float*)d_in[4];  // [L,4H,H]
  const float* b_ih   = (const float*)d_in[5];  // [L,4H]
  const float* b_hh   = (const float*)d_in[6];  // [L,4H]
  float* out = (float*)d_out;                   // output [B,S,H] ++ hx_f [L,B,H] ++ cx_f [L,B,H]

  // Workspace layout
  unsigned short* wih_pack = (unsigned short*)d_ws;                  // 16 MB
  unsigned short* whh_pack = wih_pack + (size_t)L * WPACK_LAYER;     // 16 MB
  unsigned short* xpack    = whh_pack + (size_t)L * WPACK_LAYER;     // 64 MB: [S][APACK_MAT]
  unsigned short* hpk0     = xpack + (size_t)S * APACK_MAT;          // [L][APACK_MAT]
  unsigned short* hpk1     = hpk0 + (size_t)L * APACK_MAT;           // [L][APACK_MAT]
  float* bias = (float*)(hpk1 + (size_t)L * APACK_MAT);              // [L,4H]
  float* hfin = bias + (size_t)L * 4 * H;                            // [L,B,H] latest h
  float* cbuf = hfin + (size_t)L * B * H;                            // [L,B,H] c state

  // --- precompute: pack weights + input, fuse bias, init state ---
  {
    int total = NTG_TOT * KT * 32;               // 262144 lane-fragments per weight
    int blk = 256, grd = (total + blk - 1) / blk;
    for (int l = 0; l < L; ++l) {
      lstm_pack_w<<<grd, blk, 0, stream>>>(W_ih + (size_t)l * 4 * H * E,
                                           wih_pack + (size_t)l * WPACK_LAYER);
      lstm_pack_w<<<grd, blk, 0, stream>>>(W_hh + (size_t)l * 4 * H * H,
                                           whh_pack + (size_t)l * WPACK_LAYER);
    }
    // input_: matrix s has element (m,k) at input_[(m*S + s)*E + k]
    int xtotal = S * B * (KDIM / 8);
    lstm_pack_act<<<(xtotal + 255) / 256, 256, 0, stream>>>(
        input_, xpack, (long)E, (long)S * E, S);
    // hx: matrix l has element (m,k) at hx[(l*B + m)*H + k]
    int htotal = L * B * (KDIM / 8);
    lstm_pack_act<<<(htotal + 255) / 256, 256, 0, stream>>>(
        hx, hpk0, (long)B * H, (long)H, L);
    lstm_prep_bias<<<(L * 4 * H + 255) / 256, 256, 0, stream>>>(b_ih, b_hh, bias);
    lstm_copy_c<<<(L * B * H + 255) / 256, 256, 0, stream>>>(cx, cbuf);
  }

  // --- sequential scan: 2 fused kernels per time step ---
  for (int t = 0; t < S; ++t) {
    unsigned short* hsrcp = (t & 1) ? hpk1 : hpk0;
    unsigned short* hdstp = (t & 1) ? hpk0 : hpk1;

    // layer 0: x = pre-packed input step t
    lstm_step<<<H / 16, 128, 0, stream>>>(
        xpack + (size_t)t * APACK_MAT,
        hsrcp,                                   // packed h_prev layer 0
        wih_pack, whh_pack, bias,
        cbuf, hfin,
        hdstp,                                   // packed h_new layer 0
        nullptr, 0);

    // layer 1: x = layer 0's packed h from this step; writes sequence output
    lstm_step<<<H / 16, 128, 0, stream>>>(
        hdstp,
        hsrcp + APACK_MAT,                       // packed h_prev layer 1
        wih_pack + WPACK_LAYER, whh_pack + WPACK_LAYER, bias + 4 * H,
        cbuf + (size_t)B * H, hfin + (size_t)B * H,
        hdstp + APACK_MAT,                       // packed h_new layer 1
        out + (size_t)t * H, S * H);
  }

  // Final states (latest h / c are maintained in hfin / cbuf).
  hipMemcpyAsync(out + (size_t)B * S * H, hfin,
                 sizeof(float) * (size_t)L * B * H, hipMemcpyDeviceToDevice, stream);
  hipMemcpyAsync(out + (size_t)B * S * H + (size_t)L * B * H, cbuf,
                 sizeof(float) * (size_t)L * B * H, hipMemcpyDeviceToDevice, stream);
}
